// TFXLNetMainLayer_30365418782751
// MI455X (gfx1250) — matrix operand order repaired and checked by reading, not compiled, not run on hardware
//
#include <hip/hip_runtime.h>

// ---------------------------------------------------------------------------
// Types for CDNA5 WMMA (wave32).
// ---------------------------------------------------------------------------
typedef __bf16 bf16;
typedef bf16  bf16x8 __attribute__((ext_vector_type(8)));
typedef bf16  v16bf  __attribute__((ext_vector_type(16)));
typedef float v8f    __attribute__((ext_vector_type(8)));

// Problem constants (from the reference).
#define QLEN  1024
#define BSZ   4
#define DMODEL 1024
#define NH    16
#define DH    64
#define DI    4096
#define RLEN  2048
#define SD    1152          // padded s-dim for bd_raw: 9*128, covers s in [1,1151]
#define M0    (QLEN * BSZ)  // 4096
#define MR    (RLEN * BSZ)  // 8192
#define SCALE 0.125f        // 1/sqrt(64)

__device__ inline v8f vzero8() {
  v8f z = {0.f, 0.f, 0.f, 0.f, 0.f, 0.f, 0.f, 0.f};
  return z;
}
__device__ inline v16bf make_frag(const bf16* p0, const bf16* p1) {
  bf16x8 lo = *(const bf16x8*)p0;
  bf16x8 hi = *(const bf16x8*)p1;
  return __builtin_shufflevector(lo, hi, 0, 1, 2, 3, 4, 5, 6, 7,
                                 8, 9, 10, 11, 12, 13, 14, 15);
}
__device__ inline v8f wmma_bf16(v16bf a, v16bf b, v8f c) {
  // D = A(16x32 bf16) x B(32x16 bf16) + C(16x16 f32)
  return __builtin_amdgcn_wmma_f32_16x16x32_bf16(false, a, false, b,
                                                 (short)0, c, false, false);
}

// ---------------------------------------------------------------------------
// Elementwise prep kernels.
// ---------------------------------------------------------------------------
__global__ __launch_bounds__(256) void cvt_bf16_kernel(
    const float* __restrict__ in, bf16* __restrict__ out, int n) {
  int i = blockIdx.x * 256 + threadIdx.x;
  if (i < n) out[i] = (bf16)in[i];
}

// out_bf[n][k] = in[k][n]  (in is [K][N] f32 row-major)
__global__ __launch_bounds__(256) void transpose_bf16_kernel(
    const float* __restrict__ in, bf16* __restrict__ out, int K, int N) {
  int i = blockIdx.x * 256 + threadIdx.x;
  if (i >= K * N) return;
  int k = i / N, n = i - k * N;
  out[(long long)n * K + k] = (bf16)in[i];
}

// Split QKV f32 [4096][3072] -> attention layouts (bf16) with q biases added.
__global__ __launch_bounds__(256) void split_qkv_kernel(
    const float* __restrict__ cqkv, const float* __restrict__ rwb,
    const float* __restrict__ rrb, bf16* __restrict__ qw,
    bf16* __restrict__ qr, bf16* __restrict__ kh, bf16* __restrict__ vT) {
  int idx = blockIdx.x * 256 + threadIdx.x;
  if (idx >= M0 * DMODEL) return;
  int m = idx >> 10, c = idx & 1023;
  int i = m >> 2, b = m & 3;
  int n = c >> 6, d = c & 63;
  int bn = b * NH + n;
  long long rowq = (long long)m * 3072;
  float q = cqkv[rowq + c];
  float k = cqkv[rowq + 1024 + c];
  float v = cqkv[rowq + 2048 + c];
  long long o = ((long long)bn * QLEN + i) * DH + d;
  qw[o] = (bf16)(q + rwb[c]);
  qr[o] = (bf16)(q + rrb[c]);
  kh[o] = (bf16)k;
  vT[((long long)bn * DH + d) * QLEN + i] = (bf16)v;
}

// Split k_r f32 [8192][1024] -> kr_bf [bn][2048][64]
__global__ __launch_bounds__(256) void split_kr_kernel(
    const float* __restrict__ cr, bf16* __restrict__ kr) {
  int idx = blockIdx.x * 256 + threadIdx.x;
  if (idx >= MR * DMODEL) return;
  int m = idx >> 10, c = idx & 1023;
  int j = m >> 2, b = m & 3;
  int n = c >> 6, d = c & 63;
  int bn = b * NH + n;
  kr[((long long)bn * RLEN + j) * DH + d] = (bf16)cr[idx];
}

// ef01[m][n][t] = sum_d (q[m, n*64+d] + r_s_bias[n,d]) * seg_embed[t,n,d]
__global__ __launch_bounds__(256) void ef_kernel(
    const float* __restrict__ cqkv, const float* __restrict__ rsb,
    const float* __restrict__ seg, float* __restrict__ ef01) {
  int idx = blockIdx.x * 256 + threadIdx.x;
  if (idx >= M0 * NH * 2) return;
  int t = idx & 1, n = (idx >> 1) & 15, m = idx >> 5;
  const float* qp = cqkv + (long long)m * 3072 + n * 64;
  const float* bp = rsb + n * 64;
  const float* sp = seg + (t * NH + n) * 64;
  float s = 0.f;
  for (int d = 0; d < 64; ++d) s += (qp[d] + bp[d]) * sp[d];
  ef01[idx] = s;
}

// codes[(i*1024+j)*4+b] = (mask ? 2 : 0) | (seg_diff ? 1 : 0)
__global__ __launch_bounds__(256) void code_kernel(
    const float* __restrict__ amask, const float* __restrict__ segmat,
    unsigned char* __restrict__ codes) {
  int idx = blockIdx.x * 256 + threadIdx.x;
  if (idx >= QLEN * QLEN * BSZ) return;
  unsigned char c = 0;
  if (amask[idx] > 0.5f) c |= 2;
  if (segmat[(long long)idx * 2 + 1] > 0.5f) c |= 1;
  codes[idx] = c;
}

// ---------------------------------------------------------------------------
// Generic batched bf16 WMMA GEMM:  C[M,N] = A[M,K] * B^T  (B stored [N][K]).
// 256 threads = 8 waves as 2x4; 128x128 block tile, 32-deep K steps.
// ALL dims must be multiples of the tile (128/128/32) -- no guards, so the
// inner loop is pure global_load -> ds_store / ds_load -> v_wmma with a
// 2-deep LDS double buffer (one barrier per K step, loads hidden under WMMA).
// ---------------------------------------------------------------------------
#define GBM 128
#define GBN 128
#define GBK 32
#define GLD 40  // padded LDS row stride (bf16 elems): 80B, 16B aligned

#define EPI_F32 0
#define EPI_RES_F32 2
#define EPI_BIAS_GELU_BF16 3
#define EPI_BIAS_RES_F32 4

__global__ __launch_bounds__(256) void gemm_bf16_kernel(
    const bf16* __restrict__ A, const bf16* __restrict__ B, int M, int N,
    int K, long long strideA, long long strideB, long long strideC, int mode,
    const float* __restrict__ bias, const float* __restrict__ res,
    float* __restrict__ outF, bf16* __restrict__ outB) {
  __shared__ bf16 sA[2][GBM][GLD];
  __shared__ bf16 sB[2][GBN][GLD];

  int bz = blockIdx.z;
  A += (long long)bz * strideA;
  B += (long long)bz * strideB;
  long long cOff = (long long)bz * strideC;

  int tid = threadIdx.x;
  int lane = tid & 31, wid = tid >> 5;
  int wm = wid >> 2, wn = wid & 3;  // 2 x 4 wave grid
  int lane1 = lane & 15, half = lane >> 4;
  int m0 = blockIdx.y * GBM;
  int n0 = blockIdx.x * GBN;

  // Staging map: each thread owns two rows (r, r+64) x one 8-elem chunk.
  const int sr = tid >> 2;          // 0..63
  const int sc = (tid & 3) * 8;     // 0,8,16,24
  const bf16* Ap0 = A + (long long)(m0 + sr) * K + sc;
  const bf16* Ap1 = A + (long long)(m0 + 64 + sr) * K + sc;
  const bf16* Bp0 = B + (long long)(n0 + sr) * K + sc;
  const bf16* Bp1 = B + (long long)(n0 + 64 + sr) * K + sc;

  v8f acc[4][2];
#pragma unroll
  for (int t = 0; t < 4; ++t)
#pragma unroll
    for (int u = 0; u < 2; ++u) acc[t][u] = vzero8();

  bf16x8 ra0, ra1, rb0, rb1;
  // Preload first tile and commit to buffer 0.
  ra0 = *(const bf16x8*)(Ap0);
  ra1 = *(const bf16x8*)(Ap1);
  rb0 = *(const bf16x8*)(Bp0);
  rb1 = *(const bf16x8*)(Bp1);
  *(bf16x8*)&sA[0][sr][sc] = ra0;
  *(bf16x8*)&sA[0][64 + sr][sc] = ra1;
  *(bf16x8*)&sB[0][sr][sc] = rb0;
  *(bf16x8*)&sB[0][64 + sr][sc] = rb1;

  int cur = 0;
  for (int k0 = 0; k0 < K; k0 += GBK) {
    const bool more = (k0 + GBK) < K;
    if (more) {  // issue next tile's loads before the barrier (overlap WMMA)
      ra0 = *(const bf16x8*)(Ap0 + k0 + GBK);
      ra1 = *(const bf16x8*)(Ap1 + k0 + GBK);
      rb0 = *(const bf16x8*)(Bp0 + k0 + GBK);
      rb1 = *(const bf16x8*)(Bp1 + k0 + GBK);
      if (k0 + 2 * GBK < K) {  // speculative L2 prefetch 2 steps ahead
        __builtin_prefetch(Ap0 + k0 + 2 * GBK, 0, 0);
        __builtin_prefetch(Bp0 + k0 + 2 * GBK, 0, 0);
      }
    }
    __syncthreads();

    v16bf af[4], bfr[2];
#pragma unroll
    for (int t = 0; t < 4; ++t) {
      const bf16* p = &sA[cur][wm * 64 + t * 16 + lane1][half * 8];
      af[t] = make_frag(p, p + 16);  // ISA A layout: K {h*8..+7, h*8+16..+23}
    }
#pragma unroll
    for (int u = 0; u < 2; ++u) {
      const bf16* p = &sB[cur][wn * 32 + u * 16 + lane1][half * 8];
      bfr[u] = make_frag(p, p + 16);  // ISA B layout: K {h*8..+7, h*8+16..+23}
    }
#pragma unroll
    for (int t = 0; t < 4; ++t)
#pragma unroll
      for (int u = 0; u < 2; ++u)
        acc[t][u] = wmma_bf16(af[t], bfr[u], acc[t][u]);

    if (more) {
      int nxt = cur ^ 1;
      *(bf16x8*)&sA[nxt][sr][sc] = ra0;
      *(bf16x8*)&sA[nxt][64 + sr][sc] = ra1;
      *(bf16x8*)&sB[nxt][sr][sc] = rb0;
      *(bf16x8*)&sB[nxt][64 + sr][sc] = rb1;
    }
    cur ^= 1;
  }

// Epilogue: C layout -> row m = base + v + 8*half, col n = base + lane1.
#pragma unroll
  for (int t = 0; t < 4; ++t) {
#pragma unroll
    for (int u = 0; u < 2; ++u) {
#pragma unroll
      for (int v = 0; v < 8; ++v) {
        int m = m0 + wm * 64 + t * 16 + v + 8 * half;
        int n = n0 + wn * 32 + u * 16 + lane1;
        float x = acc[t][u][v];
        long long ci = cOff + (long long)m * N + n;
        switch (mode) {
          case EPI_F32:
            outF[ci] = x;
            break;
          case EPI_RES_F32:
            outF[ci] = x + res[ci];
            break;
          case EPI_BIAS_GELU_BF16: {
            float xx = x + bias[n];
            float g = 0.5f * xx * (1.0f + erff(xx * 0.7071067811865475f));
            outB[ci] = (bf16)g;
            break;
          }
          case EPI_BIAS_RES_F32:
            outF[ci] = x + bias[n] + res[ci];
            break;
        }
      }
    }
  }
}

// ---------------------------------------------------------------------------
// Fused relative attention (flash-style): per block = 128 query rows of one
// (b, head). Each of 8 waves owns 16 rows; j sweeps causal 64-wide tiles.
// scores = (Q*K^T [wmma] + bd_raw gather + segment term) * SCALE, masked,
// streaming softmax, O += P*V [wmma].
// ---------------------------------------------------------------------------
__global__ __launch_bounds__(256) void attn_kernel(
    const bf16* __restrict__ qw,          // [bn][1024][64] (q + r_w_bias)
    const bf16* __restrict__ kh,          // [bn][1024][64]
    const bf16* __restrict__ vT,          // [bn][64][1024]
    const float* __restrict__ bd,         // [bn][1024][SD]
    const float* __restrict__ ef01,       // [m][n][2]
    const unsigned char* __restrict__ codes,  // [i][j][b]
    bf16* __restrict__ avec) {            // [m][1024]
  __shared__ bf16 sK[64][72];
  __shared__ bf16 sV[64][72];
  __shared__ bf16 sP[8][16][72];

  const int bn = blockIdx.y;
  const int b = bn >> 4, n = bn & 15;
  const int iBlock = blockIdx.x * 128;
  const int tid = threadIdx.x, lane = tid & 31, wid = tid >> 5;
  const int lane1 = lane & 15, half = lane >> 4;
  const int i0w = iBlock + wid * 16;

  // Q fragments (A layout), one per 32-deep K step over d.
  v16bf aq[2];
  {
    const bf16* qrow = qw + ((long long)bn * QLEN + i0w + lane1) * DH;
#pragma unroll
    for (int ks = 0; ks < 2; ++ks) {
      const bf16* p = qrow + ks * 32 + half * 8;
      aq[ks] = make_frag(p, p + 16);
    }
  }

  v8f oacc[4];
#pragma unroll
  for (int c = 0; c < 4; ++c) oacc[c] = vzero8();
  float mrun[8], lrun[8];
#pragma unroll
  for (int v = 0; v < 8; ++v) {
    mrun[v] = -3.0e38f;
    lrun[v] = 0.f;
  }

  const int jEnd = iBlock + 128;  // causal: j <= i < iBlock+128
  for (int j0 = 0; j0 < jEnd; j0 += 64) {
    __syncthreads();
// Stage K tile [64 j][64 d] and V^T tile [64 d][64 j].
#pragma unroll
    for (int it = 0; it < 2; ++it) {
      int idx = tid + it * 256;
      int r = idx >> 3;
      int ck = (idx & 7) * 8;
      *(bf16x8*)&sK[r][ck] =
          *(const bf16x8*)(kh + ((long long)bn * QLEN + j0 + r) * DH + ck);
      *(bf16x8*)&sV[r][ck] =
          *(const bf16x8*)(vT + ((long long)bn * DH + r) * QLEN + j0 + ck);
    }
    if (j0 + 64 < jEnd) {  // speculative prefetch of next j tile
      __builtin_prefetch(kh + ((long long)bn * QLEN + j0 + 64 + (tid >> 2)) * DH,
                         0, 0);
      __builtin_prefetch(vT + ((long long)bn * DH + (tid >> 2)) * QLEN + j0 + 64,
                         0, 0);
    }
    __syncthreads();

    // S = Q * K^T : 16x64 per wave = 4 col tiles x 2 K steps.
    v8f sacc[4];
#pragma unroll
    for (int c = 0; c < 4; ++c) sacc[c] = vzero8();
#pragma unroll
    for (int c = 0; c < 4; ++c) {
#pragma unroll
      for (int ks = 0; ks < 2; ++ks) {
        const bf16* p = &sK[c * 16 + lane1][ks * 32 + half * 8];
        v16bf bk = make_frag(p, p + 16);
        sacc[c] = wmma_bf16(aq[ks], bk, sacc[c]);
      }
    }

    // Score epilogue: + bd (rel_shift gather) + segment term, scale, mask.
    float pm[8];
#pragma unroll
    for (int v = 0; v < 8; ++v) pm[v] = -3.0e38f;
#pragma unroll
    for (int c = 0; c < 4; ++c) {
#pragma unroll
      for (int v = 0; v < 8; ++v) {
        int a = v + 8 * half;
        int i = i0w + a;
        int j = j0 + c * 16 + lane1;
        int sidx = QLEN + j - i;  // rel_shift: bd[i,j] = raw[i, j+QLEN-i]
        sidx = (sidx < 0) ? 0 : ((sidx > SD - 1) ? (SD - 1) : sidx);
        float bdv = bd[((long long)bn * QLEN + i) * SD + sidx];
        unsigned int code = codes[(i * QLEN + j) * BSZ + b];
        const float* e2 = ef01 + (((long long)(i * BSZ + b)) * NH + n) * 2;
        float ev = (code & 1) ? e2[1] : e2[0];
        float sc = (code & 2) ? -1.0e30f : (sacc[c][v] + bdv + ev) * SCALE;
        sacc[c][v] = sc;
        pm[v] = fmaxf(pm[v], sc);
      }
    }
// Row max across the 16 lanes of each half.
#pragma unroll
    for (int v = 0; v < 8; ++v) {
      float x = pm[v];
      x = fmaxf(x, __shfl_xor(x, 1));
      x = fmaxf(x, __shfl_xor(x, 2));
      x = fmaxf(x, __shfl_xor(x, 4));
      x = fmaxf(x, __shfl_xor(x, 8));
      pm[v] = x;
    }
// Streaming softmax update; P -> per-wave LDS strip (bf16, row-major).
#pragma unroll
    for (int v = 0; v < 8; ++v) {
      float mnew = fmaxf(mrun[v], pm[v]);
      float scl = __expf(mrun[v] - mnew);
      mrun[v] = mnew;
      lrun[v] *= scl;
#pragma unroll
      for (int c = 0; c < 4; ++c) oacc[c][v] *= scl;
      float rs = 0.f;
#pragma unroll
      for (int c = 0; c < 4; ++c) {
        float p = __expf(sacc[c][v] - mnew);
        rs += p;
        sP[wid][v + 8 * half][c * 16 + lane1] = (bf16)p;
      }
      rs += __shfl_xor(rs, 1);
      rs += __shfl_xor(rs, 2);
      rs += __shfl_xor(rs, 4);
      rs += __shfl_xor(rs, 8);
      lrun[v] += rs;
    }

// O += P * V : P re-read from LDS in A layout; V^T gives contiguous B frags.
#pragma unroll
    for (int ks = 0; ks < 2; ++ks) {
      const bf16* pp = &sP[wid][lane1][ks * 32 + half * 8];
      v16bf ap = make_frag(pp, pp + 16);
#pragma unroll
      for (int c = 0; c < 4; ++c) {
        const bf16* vp = &sV[c * 16 + lane1][ks * 32 + half * 8];
        v16bf bv = make_frag(vp, vp + 16);
        oacc[c] = wmma_bf16(ap, bv, oacc[c]);
      }
    }
  }

// Finalize: O /= l, write bf16 attn_vec in [m = i*B+b][n*64+d] layout.
#pragma unroll
  for (int v = 0; v < 8; ++v) {
    float inv = (lrun[v] > 0.f) ? (1.f / lrun[v]) : 0.f;
    int i = i0w + v + 8 * half;
    long long base = ((long long)i * BSZ + b) * DMODEL + n * DH;
#pragma unroll
    for (int c = 0; c < 4; ++c)
      avec[base + c * 16 + lane1] = (bf16)(oacc[c][v] * inv);
  }
}

// ---------------------------------------------------------------------------
// LayerNorm over last dim (1024); one block per row.
// ---------------------------------------------------------------------------
__global__ __launch_bounds__(256) void layernorm_kernel(
    const float* __restrict__ x, const float* __restrict__ g,
    const float* __restrict__ bb, float* __restrict__ yF,
    bf16* __restrict__ yB) {
  __shared__ float red[2][8];
  long long row = blockIdx.x;
  const float* xr = x + row * DMODEL;
  float s = 0.f, s2 = 0.f;
  for (int c = threadIdx.x; c < DMODEL; c += 256) {
    float v = xr[c];
    s += v;
    s2 += v * v;
  }
#pragma unroll
  for (int m = 1; m < 32; m <<= 1) {
    s += __shfl_xor(s, m);
    s2 += __shfl_xor(s2, m);
  }
  int wid = threadIdx.x >> 5, lane = threadIdx.x & 31;
  if (lane == 0) {
    red[0][wid] = s;
    red[1][wid] = s2;
  }
  __syncthreads();
  float ts = 0.f, ts2 = 0.f;
#pragma unroll
  for (int w = 0; w < 8; ++w) {
    ts += red[0][w];
    ts2 += red[1][w];
  }
  float mu = ts * (1.0f / DMODEL);
  float var = ts2 * (1.0f / DMODEL) - mu * mu;
  float rinv = rsqrtf(var + 1e-12f);
  for (int c = threadIdx.x; c < DMODEL; c += 256) {
    float yv = (xr[c] - mu) * rinv * g[c] + bb[c];
    yF[row * DMODEL + c] = yv;
    if (yB) yB[row * DMODEL + c] = (bf16)yv;
  }
}

// ---------------------------------------------------------------------------
// Host-side orchestration.
// ---------------------------------------------------------------------------
static inline size_t wsAlign(size_t x) { return (x + 255) & ~(size_t)255; }

extern "C" void kernel_launch(void* const* d_in, const int* in_sizes, int n_in,
                              void* d_out, int out_size, void* d_ws,
                              size_t ws_size, hipStream_t stream) {
  (void)in_sizes;
  (void)n_in;
  (void)out_size;
  (void)ws_size;
  const float* h = (const float*)d_in[0];
  const float* r = (const float*)d_in[1];
  const float* attn_mask = (const float*)d_in[2];
  const float* seg_mat = (const float*)d_in[3];
  const float* q_w = (const float*)d_in[4];
  const float* k_w = (const float*)d_in[5];
  const float* v_w = (const float*)d_in[6];
  const float* o_w = (const float*)d_in[7];
  const float* r_w = (const float*)d_in[8];
  const float* r_r_bias = (const float*)d_in[9];
  const float* r_s_bias = (const float*)d_in[10];
  const float* r_w_bias = (const float*)d_in[11];
  const float* seg_embed = (const float*)d_in[12];
  const float* ln1_g = (const float*)d_in[13];
  const float* ln1_b = (const float*)d_in[14];
  const float* ff_w1 = (const float*)d_in[15];
  const float* ff_b1 = (const float*)d_in[16];
  const float* ff_w2 = (const float*)d_in[17];
  const float* ff_b2 = (const float*)d_in[18];
  const float* ln2_g = (const float*)d_in[19];
  const float* ln2_b = (const float*)d_in[20];
  float* out = (float*)d_out;

  char* base = (char*)d_ws;
  size_t off = 0;
  auto alloc = [&](size_t bytes) -> char* {
    char* p = base + off;
    off = wsAlign(off + bytes);
    return p;
  };

  bf16* hbf = (bf16*)alloc((size_t)M0 * DMODEL * 2);
  bf16* rbf = (bf16*)alloc((size_t)MR * DMODEL * 2);
  bf16* qkvT = (bf16*)alloc((size_t)3 * DMODEL * DMODEL * 2);
  bf16* rwT = (bf16*)alloc((size_t)DMODEL * DMODEL * 2);
  bf16* owbf = (bf16*)alloc((size_t)DMODEL * DMODEL * 2);
  bf16* ff1T = (bf16*)alloc((size_t)DI * DMODEL * 2);
  bf16* ff2T = (bf16*)alloc((size_t)DMODEL * DI * 2);
  float* cqkv = (float*)alloc((size_t)M0 * 3 * DMODEL * 4);
  float* cr = (float*)alloc((size_t)MR * DMODEL * 4);
  bf16* qwb = (bf16*)alloc((size_t)BSZ * NH * QLEN * DH * 2);
  bf16* qrb = (bf16*)alloc((size_t)BSZ * NH * QLEN * DH * 2);
  bf16* khb = (bf16*)alloc((size_t)BSZ * NH * QLEN * DH * 2);
  bf16* vTb = (bf16*)alloc((size_t)BSZ * NH * DH * QLEN * 2);
  bf16* krb = (bf16*)alloc((size_t)BSZ * NH * RLEN * DH * 2);
  float* ef01 = (float*)alloc((size_t)M0 * NH * 2 * 4);
  unsigned char* codes = (unsigned char*)alloc((size_t)QLEN * QLEN * BSZ);
  float* bdws = (float*)alloc((size_t)BSZ * NH * QLEN * SD * 4);
  bf16* avec = (bf16*)alloc((size_t)M0 * DMODEL * 2);
  float* attnres = (float*)alloc((size_t)M0 * DMODEL * 4);
  float* out1 = (float*)alloc((size_t)M0 * DMODEL * 4);
  bf16* out1bf = (bf16*)alloc((size_t)M0 * DMODEL * 2);
  bf16* ffa = (bf16*)alloc((size_t)M0 * DI * 2);
  float* ff2o = (float*)alloc((size_t)M0 * DMODEL * 4);

  // ---- Prep: conversions & weight transposes ----
  cvt_bf16_kernel<<<(M0 * DMODEL + 255) / 256, 256, 0, stream>>>(h, hbf,
                                                                 M0 * DMODEL);
  cvt_bf16_kernel<<<(MR * DMODEL + 255) / 256, 256, 0, stream>>>(r, rbf,
                                                                 MR * DMODEL);
  cvt_bf16_kernel<<<(DMODEL * DMODEL + 255) / 256, 256, 0, stream>>>(
      o_w, owbf, DMODEL * DMODEL);  // o_w is already [N=h][K=nd]
  int tg = (DMODEL * DMODEL + 255) / 256;
  transpose_bf16_kernel<<<tg, 256, 0, stream>>>(q_w, qkvT + 0, DMODEL, DMODEL);
  transpose_bf16_kernel<<<tg, 256, 0, stream>>>(
      k_w, qkvT + (size_t)DMODEL * DMODEL, DMODEL, DMODEL);
  transpose_bf16_kernel<<<tg, 256, 0, stream>>>(
      v_w, qkvT + (size_t)2 * DMODEL * DMODEL, DMODEL, DMODEL);
  transpose_bf16_kernel<<<tg, 256, 0, stream>>>(r_w, rwT, DMODEL, DMODEL);
  transpose_bf16_kernel<<<(DMODEL * DI + 255) / 256, 256, 0, stream>>>(
      ff_w1, ff1T, DMODEL, DI);
  transpose_bf16_kernel<<<(DI * DMODEL + 255) / 256, 256, 0, stream>>>(
      ff_w2, ff2T, DI, DMODEL);

  // ---- Projections (WMMA) ----
  {  // cqkv[4096,3072] = hbf x qkvT^T
    dim3 g(3 * DMODEL / GBN, M0 / GBM, 1);
    gemm_bf16_kernel<<<g, 256, 0, stream>>>(hbf, qkvT, M0, 3 * DMODEL, DMODEL,
                                            0, 0, 0, EPI_F32, nullptr, nullptr,
                                            cqkv, nullptr);
  }
  {  // cr[8192,1024] = rbf x rwT^T
    dim3 g(DMODEL / GBN, MR / GBM, 1);
    gemm_bf16_kernel<<<g, 256, 0, stream>>>(rbf, rwT, MR, DMODEL, DMODEL, 0, 0,
                                            0, EPI_F32, nullptr, nullptr, cr,
                                            nullptr);
  }

  // ---- Split / small prep ----
  split_qkv_kernel<<<(M0 * DMODEL + 255) / 256, 256, 0, stream>>>(
      cqkv, r_w_bias, r_r_bias, qwb, qrb, khb, vTb);
  split_kr_kernel<<<(MR * DMODEL + 255) / 256, 256, 0, stream>>>(cr, krb);
  ef_kernel<<<(M0 * NH * 2 + 255) / 256, 256, 0, stream>>>(cqkv, r_s_bias,
                                                           seg_embed, ef01);
  code_kernel<<<(QLEN * QLEN * BSZ + 255) / 256, 256, 0, stream>>>(
      attn_mask, seg_mat, codes);

  // ---- bd_raw batched GEMM: per (b,n): [1024, SD] = qr x kr^T ----
  {
    dim3 g(SD / GBN, QLEN / GBM, BSZ * NH);
    gemm_bf16_kernel<<<g, 256, 0, stream>>>(
        qrb, krb, QLEN, SD, DH, (long long)QLEN * DH, (long long)RLEN * DH,
        (long long)QLEN * SD, EPI_F32, nullptr, nullptr, bdws, nullptr);
  }

  // ---- Fused relative attention ----
  {
    dim3 g(QLEN / 128, BSZ * NH, 1);
    attn_kernel<<<g, 256, 0, stream>>>(qwb, khb, vTb, bdws, ef01, codes, avec);
  }

  // ---- Output projection + residual ----
  {
    dim3 g(DMODEL / GBN, M0 / GBM, 1);
    gemm_bf16_kernel<<<g, 256, 0, stream>>>(avec, owbf, M0, DMODEL, DMODEL, 0,
                                            0, 0, EPI_RES_F32, nullptr, h,
                                            attnres, nullptr);
  }
  layernorm_kernel<<<M0, 256, 0, stream>>>(attnres, ln1_g, ln1_b, out1,
                                           out1bf);

  // ---- FFN ----
  {
    dim3 g(DI / GBN, M0 / GBM, 1);
    gemm_bf16_kernel<<<g, 256, 0, stream>>>(out1bf, ff1T, M0, DI, DMODEL, 0, 0,
                                            0, EPI_BIAS_GELU_BF16, ff_b1,
                                            nullptr, nullptr, ffa);
  }
  {
    dim3 g(DMODEL / GBN, M0 / GBM, 1);
    gemm_bf16_kernel<<<g, 256, 0, stream>>>(ffa, ff2T, M0, DMODEL, DI, 0, 0, 0,
                                            EPI_BIAS_RES_F32, ff_b2, out1,
                                            ff2o, nullptr);
  }
  layernorm_kernel<<<M0, 256, 0, stream>>>(ff2o, ln2_g, ln2_b, out, nullptr);
}
